// twoD_feats_Stack_22840636080142
// MI455X (gfx1250) — compile-verified, matrix operand order checked by reading, not run on hardware
//
#include <hip/hip_runtime.h>
#include <hip/hip_bf16.h>
#include <math.h>

typedef __attribute__((ext_vector_type(16))) __bf16 v16bf;
typedef __attribute__((ext_vector_type(4)))  __bf16 v4bf;
typedef __attribute__((ext_vector_type(8)))  float  v8f;

#define NPAIR 256
#define CZ    128
#define ROWS  (256*256)      // 65536
#define NH    4
#define CA    32

#define ACT_NONE 0
#define ACT_SIGMOID 1
#define ACT_RELU 2

__device__ __forceinline__ float sigf(float x) { return 1.0f / (1.0f + __expf(-x)); }

// WMMA operand swizzle (CDNA5 ISA 7.12.2, 16-bit 16xK operand layout):
//   lane = (row_or_col & 15) | (((k>>3)&1)<<4) ;  half h = (k&7) | ((k>>4)<<3)
__device__ __forceinline__ int swz_lane(int mn, int k) { return (mn & 15) | (((k >> 3) & 1) << 4); }
__device__ __forceinline__ int swz_h(int k)            { return (k & 7) | ((k >> 4) << 3); }

__device__ __forceinline__ v4bf pack4(float4 v) {
  v4bf p; p[0] = (__bf16)v.x; p[1] = (__bf16)v.y; p[2] = (__bf16)v.z; p[3] = (__bf16)v.w;
  return p;
}

// ---------------------------------------------------------------------------
// LayerNorm over last dim (128). One wave per row, 8 rows per 256-thread block.
// ---------------------------------------------------------------------------
__global__ __launch_bounds__(256) void ln_kernel(const float* __restrict__ x,
                                                 const float* __restrict__ w,
                                                 const float* __restrict__ b,
                                                 float* __restrict__ out) {
  int wv = threadIdx.x >> 5, lane = threadIdx.x & 31;
  size_t row = (size_t)blockIdx.x * 8 + wv;
  float4 v = ((const float4*)(x + row * CZ))[lane];
  float s = v.x + v.y + v.z + v.w;
#pragma unroll
  for (int o = 16; o > 0; o >>= 1) s += __shfl_xor(s, o, 32);
  float m = s * (1.0f / 128.0f);
  float d0 = v.x - m, d1 = v.y - m, d2 = v.z - m, d3 = v.w - m;
  float q = d0 * d0 + d1 * d1 + d2 * d2 + d3 * d3;
#pragma unroll
  for (int o = 16; o > 0; o >>= 1) q += __shfl_xor(q, o, 32);
  float inv = rsqrtf(q * (1.0f / 128.0f) + 1e-5f);
  float4 ww = ((const float4*)w)[lane], bb = ((const float4*)b)[lane];
  float4 r4;
  r4.x = d0 * inv * ww.x + bb.x;
  r4.y = d1 * inv * ww.y + bb.y;
  r4.z = d2 * inv * ww.z + bb.z;
  r4.w = d3 * inv * ww.w + bb.w;
  ((float4*)(out + row * CZ))[lane] = r4;
}

// ---------------------------------------------------------------------------
// GEMM: out[M,NN] = epi(A[M,KK] @ B[KK,NN]); block tile 128x128, 8 waves,
// K chunks of 32, bf16 WMMA f32 acc. Compile-time shapes + epilogue flags.
// ---------------------------------------------------------------------------
template <int NN, int KK, int ACT, bool HAS_BIAS, bool HAS_GATE, bool HAS_RES>
__global__ __launch_bounds__(256) void gemm_kernel(const float* __restrict__ A,
                                                   const float* __restrict__ B,
                                                   const float* __restrict__ bias,
                                                   const float* __restrict__ gate,
                                                   const float* __restrict__ res,
                                                   float* __restrict__ out) {
  __shared__ __align__(32) __bf16 As[8][32][16];
  __shared__ __align__(32) __bf16 Bs[8][32][16];
  int tid = threadIdx.x, wv = tid >> 5, lane = tid & 31;
  int laneHi = lane >> 4, ln15 = lane & 15;
  int m0 = blockIdx.y * 128, n0 = blockIdx.x * 128;
  v8f acc[8] = {};

  for (int k0 = 0; k0 < KK; k0 += 32) {
    __syncthreads();
#pragma unroll
    for (int s = 0; s < 4; ++s) {
      int e = tid + s * 256;                       // 0..1023 (float4 index)
      int m = e >> 3, k4 = (e & 7) << 2;           // A: 128 rows x 8 float4
      float4 av = *(const float4*)&A[(size_t)(m0 + m) * KK + k0 + k4];
      *(v4bf*)&As[m >> 4][swz_lane(m, k4)][swz_h(k4)] = pack4(av);
      int kb = e >> 5, n4 = (e & 31) << 2;         // B: 32 rows x 32 float4
      float4 bv = *(const float4*)&B[(size_t)(k0 + kb) * NN + n0 + n4];
      int hh = swz_h(kb), bt = n4 >> 4;
      Bs[bt][swz_lane(n4 + 0, kb)][hh] = (__bf16)bv.x;
      Bs[bt][swz_lane(n4 + 1, kb)][hh] = (__bf16)bv.y;
      Bs[bt][swz_lane(n4 + 2, kb)][hh] = (__bf16)bv.z;
      Bs[bt][swz_lane(n4 + 3, kb)][hh] = (__bf16)bv.w;
    }
    __syncthreads();
    v16bf a = *(const v16bf*)As[wv][lane];
#pragma unroll
    for (int nt = 0; nt < 8; ++nt) {
      v16bf bm = *(const v16bf*)Bs[nt][lane];
      acc[nt] = __builtin_amdgcn_wmma_f32_16x16x32_bf16(false, a, false, bm,
                                                        (short)0, acc[nt], false, false);
    }
  }
#pragma unroll
  for (int nt = 0; nt < 8; ++nt) {
    int n = n0 + nt * 16 + ln15;
    float bvv = HAS_BIAS ? bias[n] : 0.0f;
#pragma unroll
    for (int r = 0; r < 8; ++r) {
      int m = m0 + wv * 16 + r + laneHi * 8;
      float v = acc[nt][r] + bvv;
      if (ACT == ACT_SIGMOID) v = sigf(v);
      if (ACT == ACT_RELU) v = fmaxf(v, 0.0f);
      size_t idx = (size_t)m * NN + n;
      if (HAS_GATE) v *= gate[idx];
      if (HAS_RES)  v += res[idx];
      out[idx] = v;
    }
  }
}

// ---------------------------------------------------------------------------
// Channel-batched triangle GEMM: Tc[c] = Ah[c] @ Bh[c]^T (256x256 planes).
// Grid (2,2,128), block 256.
// ---------------------------------------------------------------------------
__global__ __launch_bounds__(256) void trigemm_kernel(const float* __restrict__ Ah,
                                                      const float* __restrict__ Bh,
                                                      float* __restrict__ Tc) {
  __shared__ __align__(32) __bf16 As[8][32][16];
  __shared__ __align__(32) __bf16 Bs[8][32][16];
  int c = blockIdx.z;
  const float* Ap = Ah + (size_t)c * ROWS;
  const float* Bp = Bh + (size_t)c * ROWS;
  float* Cp = Tc + (size_t)c * ROWS;
  int tid = threadIdx.x, wv = tid >> 5, lane = tid & 31;
  int laneHi = lane >> 4, ln15 = lane & 15;
  int m0 = blockIdx.y * 128, n0 = blockIdx.x * 128;
  v8f acc[8] = {};

  for (int k0 = 0; k0 < 256; k0 += 32) {
    __syncthreads();
#pragma unroll
    for (int s = 0; s < 4; ++s) {
      int e = tid + s * 256;                       // float4 index
      int mn = e >> 3, k4 = (e & 7) << 2;          // vectorized over k for both
      float4 av = *(const float4*)&Ap[(size_t)(m0 + mn) * 256 + k0 + k4];
      *(v4bf*)&As[mn >> 4][swz_lane(mn, k4)][swz_h(k4)] = pack4(av);
      float4 bv = *(const float4*)&Bp[(size_t)(n0 + mn) * 256 + k0 + k4];  // B^T
      *(v4bf*)&Bs[mn >> 4][swz_lane(mn, k4)][swz_h(k4)] = pack4(bv);
    }
    __syncthreads();
    v16bf a = *(const v16bf*)As[wv][lane];
#pragma unroll
    for (int nt = 0; nt < 8; ++nt) {
      v16bf bm = *(const v16bf*)Bs[nt][lane];
      acc[nt] = __builtin_amdgcn_wmma_f32_16x16x32_bf16(false, a, false, bm,
                                                        (short)0, acc[nt], false, false);
    }
  }
#pragma unroll
  for (int nt = 0; nt < 8; ++nt)
#pragma unroll
    for (int r = 0; r < 8; ++r) {
      int m = m0 + wv * 16 + r + laneHi * 8;
      int n = n0 + nt * 16 + ln15;
      Cp[(size_t)m * 256 + n] = acc[nt][r];
    }
}

// ---------------------------------------------------------------------------
// combine sigmoid-gate * proj, scatter into channel-major planes.
// trans=0: dst[c][i][k]; trans=1: dst[c][k][i]  (r = i*256+k)
// ---------------------------------------------------------------------------
__global__ __launch_bounds__(256) void combine_scatter_kernel(const float* __restrict__ sg,
                                                              const float* __restrict__ p,
                                                              float* __restrict__ dst, int trans) {
  size_t e4 = (size_t)blockIdx.x * 256 + threadIdx.x;    // float4 index over ROWS*CZ/4
  size_t idx = e4 << 2;
  int c = (int)(idx & 127);
  size_t r = idx >> 7;
  int i = (int)(r >> 8), k = (int)(r & 255);
  float4 a = ((const float4*)sg)[e4];
  float4 b = ((const float4*)p)[e4];
  size_t base = trans ? ((size_t)k * 256 + i) : ((size_t)i * 256 + k);
  dst[(size_t)(c + 0) * ROWS + base] = a.x * b.x;
  dst[(size_t)(c + 1) * ROWS + base] = a.y * b.y;
  dst[(size_t)(c + 2) * ROWS + base] = a.z * b.z;
  dst[(size_t)(c + 3) * ROWS + base] = a.w * b.w;
}

// t[r,c] = Tc[c][r]
__global__ __launch_bounds__(256) void untranspose_kernel(const float* __restrict__ Tc,
                                                          float* __restrict__ t) {
  size_t e4 = (size_t)blockIdx.x * 256 + threadIdx.x;
  size_t idx = e4 << 2;
  int c = (int)(idx & 127);
  size_t r = idx >> 7;
  float4 o;
  o.x = Tc[(size_t)(c + 0) * ROWS + r];
  o.y = Tc[(size_t)(c + 1) * ROWS + r];
  o.z = Tc[(size_t)(c + 2) * ROWS + r];
  o.w = Tc[(size_t)(c + 3) * ROWS + r];
  ((float4*)t)[e4] = o;
}

// zt[j,i,:] = z[i,j,:]
__global__ __launch_bounds__(256) void transpose_z_kernel(const float* __restrict__ in,
                                                          float* __restrict__ out) {
  size_t e4 = (size_t)blockIdx.x * 256 + threadIdx.x;
  size_t idx = e4 << 2;
  int c = (int)(idx & 127);
  size_t r = idx >> 7;
  int i = (int)(r >> 8), j = (int)(r & 255);
  float4 v = ((const float4*)in)[e4];
  *(float4*)&out[((size_t)j * 256 + i) * CZ + c] = v;
}

// bias[r,h] = sum_c x[r,c] * bw[c,h]   (H=4). One wave per row.
__global__ __launch_bounds__(256) void biasproj_kernel(const float* __restrict__ x,
                                                       const float* __restrict__ bw,
                                                       float* __restrict__ biasb) {
  int wv = threadIdx.x >> 5, lane = threadIdx.x & 31;
  size_t row = (size_t)blockIdx.x * 8 + wv;
  float4 xv = ((const float4*)(x + row * CZ))[lane];
  const float4* bwv = (const float4*)bw;
  float4 b0 = bwv[lane * 4 + 0], b1 = bwv[lane * 4 + 1];
  float4 b2 = bwv[lane * 4 + 2], b3 = bwv[lane * 4 + 3];
  float4 a;
  a.x = xv.x * b0.x + xv.y * b1.x + xv.z * b2.x + xv.w * b3.x;
  a.y = xv.x * b0.y + xv.y * b1.y + xv.z * b2.y + xv.w * b3.y;
  a.z = xv.x * b0.z + xv.y * b1.z + xv.z * b2.z + xv.w * b3.z;
  a.w = xv.x * b0.w + xv.y * b1.w + xv.z * b2.w + xv.w * b3.w;
#pragma unroll
  for (int o = 16; o > 0; o >>= 1) {
    a.x += __shfl_xor(a.x, o, 32);
    a.y += __shfl_xor(a.y, o, 32);
    a.z += __shfl_xor(a.z, o, 32);
    a.w += __shfl_xor(a.w, o, 32);
  }
  if (lane == 0) ((float4*)(biasb + row * 4))[0] = a;
}

// ---------------------------------------------------------------------------
// Triangle attention: block = (32 q-rows, head h, row i); 2 waves.
// scores = (Q*scale) K^T + bias ; softmax(256) ; O = P V ; O *= gate.
// ---------------------------------------------------------------------------
__global__ __launch_bounds__(64) void attn_kernel(const float* __restrict__ qb,
                                                  const float* __restrict__ kb,
                                                  const float* __restrict__ vb,
                                                  const float* __restrict__ biasb,
                                                  const float* __restrict__ gateb,
                                                  float* __restrict__ ob) {
  __shared__ float sc[32][256];
  __shared__ __align__(32) __bf16 Ks[16][32][16];
  __shared__ __align__(32) __bf16 Vs[8][2][32][16];
  int qc = blockIdx.x, h = blockIdx.y, i = blockIdx.z;
  int tid = threadIdx.x, wv = tid >> 5, lane = tid & 31;
  int laneHi = lane >> 4, ln15 = lane & 15;
  const float scale = 0.17677669529663689f;          // 1/sqrt(32)
  size_t rowbase = (size_t)i * 256;

  // stage K and V (256 x 32 each), float4 over channel
#pragma unroll 4
  for (int s = 0; s < 32; ++s) {
    int e = tid + s * 64;                            // float4 index, 0..2047
    int kj = e >> 3, c4 = (e & 7) << 2;
    size_t g = (rowbase + kj) * CZ + h * CA + c4;
    float4 kv = *(const float4*)&kb[g];
    *(v4bf*)&Ks[kj >> 4][swz_lane(kj, c4)][swz_h(c4)] = pack4(kv);
    float4 vv = *(const float4*)&vb[g];
    int kl = kj & 31, hh = swz_h(kl), kc = kj >> 5, nt = c4 >> 4;
    Vs[kc][nt][swz_lane(c4 + 0, kl)][hh] = (__bf16)vv.x;
    Vs[kc][nt][swz_lane(c4 + 1, kl)][hh] = (__bf16)vv.y;
    Vs[kc][nt][swz_lane(c4 + 2, kl)][hh] = (__bf16)vv.z;
    Vs[kc][nt][swz_lane(c4 + 3, kl)][hh] = (__bf16)vv.w;
  }

  // Q tile (16 rows x 32 ch) -> WMMA A layout: two runs of 8 consecutive k
  int q0 = qc * 32 + wv * 16;
  const float* qrow = &qb[(rowbase + q0 + ln15) * CZ + h * CA + laneHi * 8];
  float4 qv0 = *(const float4*)qrow;
  float4 qv1 = *(const float4*)(qrow + 4);
  float4 qv2 = *(const float4*)(qrow + 16);
  float4 qv3 = *(const float4*)(qrow + 20);
  v16bf qa;
  qa[0] = (__bf16)(qv0.x * scale); qa[1] = (__bf16)(qv0.y * scale);
  qa[2] = (__bf16)(qv0.z * scale); qa[3] = (__bf16)(qv0.w * scale);
  qa[4] = (__bf16)(qv1.x * scale); qa[5] = (__bf16)(qv1.y * scale);
  qa[6] = (__bf16)(qv1.z * scale); qa[7] = (__bf16)(qv1.w * scale);
  qa[8]  = (__bf16)(qv2.x * scale); qa[9]  = (__bf16)(qv2.y * scale);
  qa[10] = (__bf16)(qv2.z * scale); qa[11] = (__bf16)(qv2.w * scale);
  qa[12] = (__bf16)(qv3.x * scale); qa[13] = (__bf16)(qv3.y * scale);
  qa[14] = (__bf16)(qv3.z * scale); qa[15] = (__bf16)(qv3.w * scale);
  __syncthreads();

  // scores + bias
#pragma unroll 2
  for (int kt = 0; kt < 16; ++kt) {
    v16bf bm = *(const v16bf*)Ks[kt][lane];
    v8f a0 = {};
    a0 = __builtin_amdgcn_wmma_f32_16x16x32_bf16(false, qa, false, bm, (short)0, a0, false, false);
#pragma unroll
    for (int r = 0; r < 8; ++r) {
      int m = r + laneHi * 8;
      int kj = kt * 16 + ln15;
      sc[wv * 16 + m][kj] = a0[r] + biasb[((size_t)(q0 + m) * 256 + kj) * 4 + h];
    }
  }
  __syncthreads();

  // softmax: 2 threads per row (paired lanes), 32 rows
  {
    int row = tid >> 1, half = tid & 1;
    float* p = &sc[row][half * 128];
    float mx = -1e30f;
    for (int j = 0; j < 128; ++j) mx = fmaxf(mx, p[j]);
    mx = fmaxf(mx, __shfl_xor(mx, 1, 32));
    float sum = 0.0f;
    for (int j = 0; j < 128; ++j) { float e_ = __expf(p[j] - mx); p[j] = e_; sum += e_; }
    sum += __shfl_xor(sum, 1, 32);
    float inv = 1.0f / sum;
    for (int j = 0; j < 128; ++j) p[j] *= inv;
  }
  __syncthreads();

  // O = P @ V  (K = 256 in 8 chunks of 32)
  v8f o0 = {}, o1 = {};
#pragma unroll 2
  for (int kc = 0; kc < 8; ++kc) {
    const float* prow = &sc[wv * 16 + ln15][kc * 32 + laneHi * 8];
    float4 p0 = *(const float4*)prow;
    float4 p1 = *(const float4*)(prow + 4);
    float4 p2 = *(const float4*)(prow + 16);
    float4 p3 = *(const float4*)(prow + 20);
    v16bf pa;
    pa[0] = (__bf16)p0.x; pa[1] = (__bf16)p0.y; pa[2] = (__bf16)p0.z; pa[3] = (__bf16)p0.w;
    pa[4] = (__bf16)p1.x; pa[5] = (__bf16)p1.y; pa[6] = (__bf16)p1.z; pa[7] = (__bf16)p1.w;
    pa[8]  = (__bf16)p2.x; pa[9]  = (__bf16)p2.y; pa[10] = (__bf16)p2.z; pa[11] = (__bf16)p2.w;
    pa[12] = (__bf16)p3.x; pa[13] = (__bf16)p3.y; pa[14] = (__bf16)p3.z; pa[15] = (__bf16)p3.w;
    v16bf b0 = *(const v16bf*)Vs[kc][0][lane];
    v16bf b1 = *(const v16bf*)Vs[kc][1][lane];
    o0 = __builtin_amdgcn_wmma_f32_16x16x32_bf16(false, pa, false, b0, (short)0, o0, false, false);
    o1 = __builtin_amdgcn_wmma_f32_16x16x32_bf16(false, pa, false, b1, (short)0, o1, false, false);
  }
#pragma unroll
  for (int r = 0; r < 8; ++r) {
    int m = r + laneHi * 8;
    size_t rr = rowbase + q0 + m;
    size_t g0 = rr * CZ + h * CA + ln15;
    size_t g1 = g0 + 16;
    ob[g0] = o0[r] * gateb[g0];
    ob[g1] = o1[r] * gateb[g1];
  }
}

// ---------------------------------------------------------------------------
// Host orchestration
// ---------------------------------------------------------------------------
extern "C" void kernel_launch(void* const* d_in, const int* in_sizes, int n_in,
                              void* d_out, int out_size, void* d_ws, size_t ws_size,
                              hipStream_t stream) {
  (void)in_sizes; (void)n_in; (void)out_size; (void)ws_size;
  auto F = [&](int i) { return (const float*)d_in[i]; };
  const float* z      = F(0);
  const float* tmlnw  = F(1);  const float* tmlnb  = F(2);
  const float* apw    = F(3);  const float* apb    = F(4);
  const float* agw    = F(5);  const float* agb    = F(6);
  const float* bpw    = F(7);  const float* bpb    = F(8);
  const float* bgw    = F(9);  const float* bgb    = F(10);
  const float* gw     = F(11); const float* gb     = F(12);
  const float* zow    = F(13); const float* zob    = F(14);
  const float* lnow   = F(15); const float* lnob   = F(16);
  const float* talnw  = F(17); const float* talnb  = F(18);
  const float* qw     = F(19); const float* kw     = F(20); const float* vw = F(21);
  const float* bw     = F(22);
  const float* tgw    = F(23); const float* tgb    = F(24);
  const float* ow     = F(25); const float* obb    = F(26);
  const float* ptlnw  = F(27); const float* ptlnb  = F(28);
  const float* pt1w   = F(29); const float* pt1b   = F(30);
  const float* pt2w   = F(31); const float* pt2b   = F(32);

  float* ws = (float*)d_ws;
  const size_t SZ = (size_t)ROWS * CZ;
  auto slot = [&](int i) { return ws + (size_t)i * SZ; };
  float* xln   = slot(0);
  float* t1    = slot(1);
  float* t2    = slot(2);
  float* Ah    = slot(3);
  float* Bh    = slot(4);
  float* Tc    = slot(5);
  float* gateb = slot(6);
  float* tbuf  = slot(7);
  float* ztb   = slot(8);
  float* zcur  = slot(9);
  float* biasb = ws + (size_t)10 * SZ;
  float* h1    = slot(1);                       // PT hidden spans slots 1..4

  const dim3 B256(256), B64(64);
  const dim3 Gln(ROWS / 8);
  const dim3 Gv4(ROWS * CZ / 1024);             // float4 elementwise grids
  const dim3 Gg128(1, ROWS / 128);
  const dim3 Gg512(4, ROWS / 128);
  const dim3 Gtri(2, 2, 128);
  const dim3 Gattn(8, NH, 256);

  hipMemcpyAsync(zcur, z, SZ * sizeof(float), hipMemcpyDeviceToDevice, stream);

  auto tri_mul = [&](int d, int outgoing) {
    size_t W = (size_t)d * CZ * CZ, V = (size_t)d * CZ;
    ln_kernel<<<Gln, B256, 0, stream>>>(zcur, tmlnw + V, tmlnb + V, xln);
    gemm_kernel<CZ, CZ, ACT_SIGMOID, true, false, false>
        <<<Gg128, B256, 0, stream>>>(xln, agw + W, agb + V, nullptr, nullptr, t1);
    gemm_kernel<CZ, CZ, ACT_NONE, true, false, false>
        <<<Gg128, B256, 0, stream>>>(xln, apw + W, apb + V, nullptr, nullptr, t2);
    combine_scatter_kernel<<<Gv4, B256, 0, stream>>>(t1, t2, Ah, outgoing ? 0 : 1);
    gemm_kernel<CZ, CZ, ACT_SIGMOID, true, false, false>
        <<<Gg128, B256, 0, stream>>>(xln, bgw + W, bgb + V, nullptr, nullptr, t1);
    gemm_kernel<CZ, CZ, ACT_NONE, true, false, false>
        <<<Gg128, B256, 0, stream>>>(xln, bpw + W, bpb + V, nullptr, nullptr, t2);
    combine_scatter_kernel<<<Gv4, B256, 0, stream>>>(t1, t2, Bh, outgoing ? 0 : 1);
    gemm_kernel<CZ, CZ, ACT_SIGMOID, true, false, false>
        <<<Gg128, B256, 0, stream>>>(xln, gw + W, gb + V, nullptr, nullptr, gateb);
    trigemm_kernel<<<Gtri, B256, 0, stream>>>(Ah, Bh, Tc);
    untranspose_kernel<<<Gv4, B256, 0, stream>>>(Tc, tbuf);
    ln_kernel<<<Gln, B256, 0, stream>>>(tbuf, lnow + V, lnob + V, t1);
    gemm_kernel<CZ, CZ, ACT_NONE, true, true, true>
        <<<Gg128, B256, 0, stream>>>(t1, zow + W, zob + V, gateb, zcur, zcur);
  };

  auto tri_att = [&](int d, float* zb) {
    size_t W = (size_t)d * CZ * CZ, V = (size_t)d * CZ, WB = (size_t)d * CZ * NH;
    ln_kernel<<<Gln, B256, 0, stream>>>(zb, talnw + V, talnb + V, xln);
    gemm_kernel<CZ, CZ, ACT_NONE, false, false, false>
        <<<Gg128, B256, 0, stream>>>(xln, qw + W, nullptr, nullptr, nullptr, t1);
    gemm_kernel<CZ, CZ, ACT_NONE, false, false, false>
        <<<Gg128, B256, 0, stream>>>(xln, kw + W, nullptr, nullptr, nullptr, t2);
    gemm_kernel<CZ, CZ, ACT_NONE, false, false, false>
        <<<Gg128, B256, 0, stream>>>(xln, vw + W, nullptr, nullptr, nullptr, tbuf);
    gemm_kernel<CZ, CZ, ACT_SIGMOID, true, false, false>
        <<<Gg128, B256, 0, stream>>>(xln, tgw + W, tgb + V, nullptr, nullptr, gateb);
    biasproj_kernel<<<Gln, B256, 0, stream>>>(xln, bw + WB, biasb);
    attn_kernel<<<Gattn, B64, 0, stream>>>(t1, t2, tbuf, biasb, gateb, Ah);
    gemm_kernel<CZ, CZ, ACT_NONE, true, false, true>
        <<<Gg128, B256, 0, stream>>>(Ah, ow + W, obb + V, nullptr, zb, zb);
  };

  tri_mul(0, 1);                                // outgoing
  tri_mul(1, 0);                                // incoming
  tri_att(0, zcur);                             // starting node
  transpose_z_kernel<<<Gv4, B256, 0, stream>>>(zcur, ztb);
  tri_att(1, ztb);                              // ending node (on z^T)
  transpose_z_kernel<<<Gv4, B256, 0, stream>>>(ztb, zcur);

  // pair transition
  ln_kernel<<<Gln, B256, 0, stream>>>(zcur, ptlnw, ptlnb, xln);
  gemm_kernel<4 * CZ, CZ, ACT_RELU, true, false, false>
      <<<Gg512, B256, 0, stream>>>(xln, pt1w, pt1b, nullptr, nullptr, h1);
  gemm_kernel<CZ, 4 * CZ, ACT_NONE, true, false, true>
      <<<Gg128, B256, 0, stream>>>(h1, pt2w, pt2b, nullptr, zcur, (float*)d_out);
}
